// VarianceAdaptor_57732950392964
// MI455X (gfx1250) — matrix operand order repaired and checked by reading, not compiled, not run on hardware
//
#include <hip/hip_runtime.h>

// ---------------- problem constants ----------------
#define B_    16
#define L_    512
#define C_    384
#define F_    384
#define K_    3
#define MEL_  4096

typedef __bf16 bf16;
typedef __attribute__((ext_vector_type(16))) __bf16 v16bf;
typedef __attribute__((ext_vector_type(8)))  __bf16 v8bf;
typedef __attribute__((ext_vector_type(4)))  __bf16 v4bf;
typedef __attribute__((ext_vector_type(8)))  float  v8f;

// ---------------- tiling ----------------
#define TBLK    32            // output timesteps per workgroup
#define XROWS   36            // TBLK + 4 (conv1 input halo incl. conv2's halo)
#define HROWS   34            // TBLK + 2 (conv1 output rows incl. conv2 halo)
#define XSTRIDE 392           // 384 + 8 bf16 pad -> conflict-free A loads
#define NWAVES  8
#define NTPW    3             // 24 N-tiles (384/16) / 8 waves

// =====================================================================
// weight prep: w[F][C][K] fp32  ->  wt[K][F][C] bf16  (B-matrix friendly)
// =====================================================================
__global__ void prep_weights_kernel(const float* __restrict__ w,
                                    bf16* __restrict__ wt) {
  int i = blockIdx.x * blockDim.x + threadIdx.x;
  if (i >= F_ * C_ * K_) return;
  int k = i % K_;
  int c = (i / K_) % C_;
  int f = i / (K_ * C_);
  wt[(k * F_ + f) * C_ + c] = (bf16)w[i];
}

// =====================================================================
// cumsum of durations (tiny)
// =====================================================================
__global__ void cumsum_kernel(const int* __restrict__ dur, int* __restrict__ cum) {
  int b = blockIdx.x;
  if (threadIdx.x == 0) {
    int s = 0;
    for (int i = 0; i < L_; ++i) { s += dur[b * L_ + i]; cum[b * L_ + i] = s; }
  }
}

// =====================================================================
// length regulate: gather + mask, writes expanded straight into d_out
// =====================================================================
__global__ void regulate_kernel(const float* __restrict__ x,
                                const int* __restrict__ cum,
                                float* __restrict__ out) {
  int b = blockIdx.y, t = blockIdx.x;
  const int* cb = cum + b * L_;
  int total = cb[L_ - 1];
  int lo = 0, hi = L_;
  while (lo < hi) { int mid = (lo + hi) >> 1; if (cb[mid] <= t) lo = mid + 1; else hi = mid; }
  int idx = (lo < L_ - 1) ? lo : (L_ - 1);
  float m = (t < total) ? 1.f : 0.f;
  const float4* src = (const float4*)(x + ((size_t)b * L_ + idx) * C_);
  float4* dst = (float4*)(out + ((size_t)b * MEL_ + t) * C_);
  for (int c = threadIdx.x; c < C_ / 4; c += blockDim.x) {
    float4 v = src[c];
    dst[c] = make_float4(v.x * m, v.y * m, v.z * m, v.w * m);
  }
}

// =====================================================================
// out[b,t,:] += p[b,t] + e[b,t]  (single fused broadcast-add pass)
// =====================================================================
__global__ void add2_kernel(float* __restrict__ out,
                            const float* __restrict__ p,
                            const float* __restrict__ e) {
  int b = blockIdx.y, t = blockIdx.x;
  float v = p[(size_t)b * MEL_ + t] + e[(size_t)b * MEL_ + t];
  float* q = out + ((size_t)b * MEL_ + t) * C_;
  for (int c = threadIdx.x; c < C_; c += blockDim.x) q[c] += v;
}

// =====================================================================
// fused predictor building blocks
// =====================================================================
__device__ __forceinline__ v16bf load_a_frag(const bf16* p) {
  union { v16bf v; v8bf h[2]; } u;
  u.h[0] = *(const v8bf*)(p);        // K chunk [c0+half*16, +8)
  u.h[1] = *(const v8bf*)(p + 8);    // K chunk [c0+half*16+8, +8)
  return u.v;
}

// conv as 3 shifted GEMMs, bf16 WMMA with f32 accumulation.
// h-row r uses A rows r+k (k=0..2). Output: bias+ReLU -> bf16.
template <int MT>
__device__ __forceinline__ void conv_gemm(const bf16* __restrict__ Abuf,
                                          const bf16* __restrict__ wt,
                                          const float* __restrict__ bias,
                                          bf16* __restrict__ Obuf,
                                          int rows_valid, int lane, int wave) {
  const int nsub  = lane & 15;
  const int khalf = lane >> 4;
  const int fbase = wave * NTPW * 16 + nsub;
  v8f acc[NTPW][MT];
  const v8f vz = {0.f, 0.f, 0.f, 0.f, 0.f, 0.f, 0.f, 0.f};
#pragma unroll
  for (int nt = 0; nt < NTPW; ++nt)
#pragma unroll
    for (int mt = 0; mt < MT; ++mt) acc[nt][mt] = vz;

#pragma unroll
  for (int k = 0; k < 3; ++k) {
#pragma unroll 2
    for (int c0 = 0; c0 < C_; c0 += 32) {
      // issue all fragment loads for this K-step up front (A: LDS, B: global)
      v16bf af[MT];
      v16bf bfr[NTPW];
#pragma unroll
      for (int mt = 0; mt < MT; ++mt)
        af[mt] = load_a_frag(Abuf + (mt * 16 + nsub + k) * XSTRIDE + c0 + khalf * 16);
#pragma unroll
      for (int nt = 0; nt < NTPW; ++nt)
        bfr[nt] = *(const v16bf*)(wt + (k * F_ + fbase + nt * 16) * C_ + c0 + khalf * 16);
#pragma unroll
      for (int nt = 0; nt < NTPW; ++nt)
#pragma unroll
        for (int mt = 0; mt < MT; ++mt)
          acc[nt][mt] = __builtin_amdgcn_wmma_f32_16x16x32_bf16(
              false, af[mt], false, bfr[nt], (short)0, acc[nt][mt], false, false);
    }
  }
  __syncthreads();  // all waves done reading Abuf (may alias Obuf)
#pragma unroll
  for (int nt = 0; nt < NTPW; ++nt) {
    int n = (wave * NTPW + nt) * 16 + nsub;
    float bv = bias[n];
#pragma unroll
    for (int mt = 0; mt < MT; ++mt)
#pragma unroll
      for (int v = 0; v < 8; ++v) {
        int r = mt * 16 + khalf * 8 + v;
        if (r < rows_valid) {
          float val = acc[nt][mt][v] + bv;
          Obuf[r * XSTRIDE + n] = (bf16)fmaxf(val, 0.f);
        }
      }
  }
  __syncthreads();
}

__device__ __forceinline__ void layernorm_rows(bf16* __restrict__ buf, int rows,
                                               const float* __restrict__ g,
                                               const float* __restrict__ be,
                                               int lane, int wave) {
  float gc[12], bc[12];
#pragma unroll
  for (int j = 0; j < 12; ++j) { gc[j] = g[lane + 32 * j]; bc[j] = be[lane + 32 * j]; }
  for (int r = wave; r < rows; r += NWAVES) {
    float s1 = 0.f, s2 = 0.f, xv[12];
#pragma unroll
    for (int j = 0; j < 12; ++j) {
      float v = (float)buf[r * XSTRIDE + lane + 32 * j];
      xv[j] = v; s1 += v; s2 += v * v;
    }
#pragma unroll
    for (int m = 16; m >= 1; m >>= 1) {
      s1 += __shfl_xor(s1, m, 32);
      s2 += __shfl_xor(s2, m, 32);
    }
    float mu  = s1 * (1.f / 384.f);
    float var = s2 * (1.f / 384.f) - mu * mu;
    float rs  = rsqrtf(var + 1e-5f);
#pragma unroll
    for (int j = 0; j < 12; ++j)
      buf[r * XSTRIDE + lane + 32 * j] = (bf16)((xv[j] - mu) * rs * gc[j] + bc[j]);
  }
  __syncthreads();
}

__device__ __forceinline__ void linear_out(const bf16* __restrict__ buf,
                                           const float* __restrict__ wl, float blv,
                                           float* __restrict__ sout, int rows,
                                           int lane, int wave) {
  float wc[12];
#pragma unroll
  for (int j = 0; j < 12; ++j) wc[j] = wl[lane + 32 * j];
  for (int r = wave; r < rows; r += NWAVES) {
    float s = 0.f;
#pragma unroll
    for (int j = 0; j < 12; ++j)
      s += (float)buf[r * XSTRIDE + lane + 32 * j] * wc[j];
#pragma unroll
    for (int m = 16; m >= 1; m >>= 1) s += __shfl_xor(s, m, 32);
    if (lane == 0) sout[r] = s + blv;
  }
}

// =====================================================================
// fused predictor: Conv->ReLU->LN->Conv->ReLU->LN->Linear per 32-row block
// add_s (nullable): per-row scalar added to the input during staging
// =====================================================================
__global__ __launch_bounds__(256, 1)
void predictor_kernel(const float* __restrict__ in, int T,
                      const float* __restrict__ add_s,
                      const bf16* __restrict__ wt1, const float* __restrict__ b1,
                      const float* __restrict__ g1, const float* __restrict__ be1,
                      const bf16* __restrict__ wt2, const float* __restrict__ b2,
                      const float* __restrict__ g2, const float* __restrict__ be2,
                      const float* __restrict__ wl, const float* __restrict__ bl,
                      float* __restrict__ sout) {
  __shared__ bf16 xs[XROWS * XSTRIDE];  // input tile (bf16) with halo
  __shared__ bf16 h1[HROWS * XSTRIDE];  // conv1 out -> LN -> conv2 out -> LN

  int tid = threadIdx.x;
  int lane = tid & 31, wave = tid >> 5;
  int b  = blockIdx.y;
  int t0 = blockIdx.x * TBLK;
  const float* xb = in + (size_t)b * T * C_;

  // stage x block [t0-2, t0+TBLK+2) into LDS as bf16 (+optional row scalar)
  const int nvec = XROWS * (C_ / 4);
  for (int i = tid; i < nvec; i += 256) {
    int r = i / (C_ / 4);
    int c = (i % (C_ / 4)) * 4;
    int gr = t0 - 2 + r;
    float4 v = make_float4(0.f, 0.f, 0.f, 0.f);
    if (gr >= 0 && gr < T) {
      v = ((const float4*)(xb + (size_t)gr * C_))[c / 4];
      if (add_s) {
        float sa = add_s[(size_t)b * T + gr];
        v.x += sa; v.y += sa; v.z += sa; v.w += sa;
      }
    }
    v4bf o = {(bf16)v.x, (bf16)v.y, (bf16)v.z, (bf16)v.w};
    *(v4bf*)(xs + r * XSTRIDE + c) = o;
  }
  __syncthreads();

  conv_gemm<3>(xs, wt1, b1, h1, HROWS, lane, wave);   // 34 rows (+1 halo each side)
  layernorm_rows(h1, HROWS, g1, be1, lane, wave);
  conv_gemm<2>(h1, wt2, b2, h1, TBLK, lane, wave);    // 32 rows, in-place (barriered)
  layernorm_rows(h1, TBLK, g2, be2, lane, wave);
  linear_out(h1, wl, bl[0], sout + (size_t)b * T + t0, TBLK, lane, wave);
}

// =====================================================================
// host-side orchestration (graph-capture safe: kernels on `stream` only)
// =====================================================================
extern "C" void kernel_launch(void* const* d_in, const int* in_sizes, int n_in,
                              void* d_out, int out_size, void* d_ws, size_t ws_size,
                              hipStream_t stream) {
  (void)in_sizes; (void)n_in; (void)out_size; (void)ws_size;

  const float* x = (const float*)d_in[0];
  // predictor params: base+0..9 = w1,b1,g1,be1,w2,b2,g2,be2,wl,bl
  const float* dp[10]; const float* pp[10]; const float* ep[10];
  for (int i = 0; i < 10; ++i) {
    dp[i] = (const float*)d_in[1 + i];
    pp[i] = (const float*)d_in[11 + i];
    ep[i] = (const float*)d_in[21 + i];
  }
  const int* dur = (const int*)d_in[31];

  // ---- workspace layout ----
  const size_t WT_ELEMS = (size_t)K_ * F_ * C_;  // 442368
  bf16* wtb = (bf16*)d_ws;
  bf16* wt_dp1 = wtb + 0 * WT_ELEMS;
  bf16* wt_dp2 = wtb + 1 * WT_ELEMS;
  bf16* wt_pp1 = wtb + 2 * WT_ELEMS;
  bf16* wt_pp2 = wtb + 3 * WT_ELEMS;
  bf16* wt_ep1 = wtb + 4 * WT_ELEMS;
  bf16* wt_ep2 = wtb + 5 * WT_ELEMS;
  int* cum = (int*)(wtb + 6 * WT_ELEMS);

  // ---- output layout: out | durations | pitches | energies ----
  float* out_main   = (float*)d_out;
  float* out_dur    = out_main + (size_t)B_ * MEL_ * C_;
  float* out_pitch  = out_dur + (size_t)B_ * L_;
  float* out_energy = out_pitch + (size_t)B_ * MEL_;

  // 1) weight prep (fp32 [F,C,K] -> bf16 [K,F,C])
  int wblocks = (int)((WT_ELEMS + 255) / 256);
  prep_weights_kernel<<<wblocks, 256, 0, stream>>>(dp[0], wt_dp1);
  prep_weights_kernel<<<wblocks, 256, 0, stream>>>(dp[4], wt_dp2);
  prep_weights_kernel<<<wblocks, 256, 0, stream>>>(pp[0], wt_pp1);
  prep_weights_kernel<<<wblocks, 256, 0, stream>>>(pp[4], wt_pp2);
  prep_weights_kernel<<<wblocks, 256, 0, stream>>>(ep[0], wt_ep1);
  prep_weights_kernel<<<wblocks, 256, 0, stream>>>(ep[4], wt_ep2);

  // 2) cumsum + length regulate -> expanded into out_main
  cumsum_kernel<<<B_, 32, 0, stream>>>(dur, cum);
  regulate_kernel<<<dim3(MEL_, B_), 96, 0, stream>>>(x, cum, out_main);

  // 3) duration predictor on x
  predictor_kernel<<<dim3(L_ / TBLK, B_), 256, 0, stream>>>(
      x, L_, nullptr, wt_dp1, dp[1], dp[2], dp[3], wt_dp2, dp[5], dp[6], dp[7],
      dp[8], dp[9], out_dur);

  // 4) pitch predictor on expanded
  predictor_kernel<<<dim3(MEL_ / TBLK, B_), 256, 0, stream>>>(
      out_main, MEL_, nullptr, wt_pp1, pp[1], pp[2], pp[3], wt_pp2, pp[5], pp[6],
      pp[7], pp[8], pp[9], out_pitch);

  // 5) energy predictor on (expanded + pitch), pitch folded into staging
  predictor_kernel<<<dim3(MEL_ / TBLK, B_), 256, 0, stream>>>(
      out_main, MEL_, out_pitch, wt_ep1, ep[1], ep[2], ep[3], wt_ep2, ep[5],
      ep[6], ep[7], ep[8], ep[9], out_energy);

  // 6) single fused pass: out = expanded + pitch + energy
  add2_kernel<<<dim3(MEL_, B_), 128, 0, stream>>>(out_main, out_pitch, out_energy);
}
